// HybridSASRecGNN_68564857913782
// MI455X (gfx1250) — matrix operand order repaired and checked by reading, not compiled, not run on hardware
//
#include <hip/hip_runtime.h>
#include <hip/hip_bf16.h>

#define B_   64
#define S_   50
#define T_   (B_ * S_)     // 3200 tokens
#define D_   64
#define H_   2
#define DH_  32
#define DFF_ 256

typedef __attribute__((ext_vector_type(16))) _Float16 v16h;
typedef __attribute__((ext_vector_type(8)))  float    v8f;

// ---------------------------------------------------------------- WMMA core
__device__ __forceinline__ v8f wmma16x16x32(v16h a, v16h b, v8f c) {
  // 8 args: (neg_a, A, neg_b, B, c_mod, C, reuse_a, reuse_b)
  return __builtin_amdgcn_wmma_f32_16x16x32_f16(false, a, false, b, (short)0, c,
                                                false, false);
}

__device__ __forceinline__ float wave_sum32(float v) {
#pragma unroll
  for (int m = 16; m >= 1; m >>= 1) v += __shfl_xor(v, m, 32);
  return v;
}

// ---------------------------------------------------------------- GNN scatter
// One edge handled by 16 threads, each doing a float4 slice of D=64.
// Node data is L2-resident (25.6 MB << 192 MB L2), edges stream.
__global__ void gnn_scatter(const int* __restrict__ src, const int* __restrict__ dst,
                            const float* __restrict__ w, const float* __restrict__ xin,
                            float* __restrict__ xout, int E) {
  long long tid = (long long)blockIdx.x * blockDim.x + threadIdx.x;
  long long e = tid >> 4;
  if (e >= E) return;
  int j = ((int)tid & 15) << 2;
  int s = src[e], d = dst[e];
  float ww = w[e];
  const float4 xv = *reinterpret_cast<const float4*>(xin + (size_t)s * D_ + j);
  float* o = xout + (size_t)d * D_ + j;
  atomicAdd(o + 0, xv.x * ww);
  atomicAdd(o + 1, xv.y * ww);
  atomicAdd(o + 2, xv.z * ww);
  atomicAdd(o + 3, xv.w * ww);
}

// ---------------------------------------------------------------- gather acc at seq tokens
__global__ void gather_acc(const int* __restrict__ seq, const float* __restrict__ emb,
                           const float* __restrict__ x1, const float* __restrict__ x2,
                           float* __restrict__ g) {
  int tid = blockIdx.x * blockDim.x + threadIdx.x;
  if (tid >= T_ * D_) return;
  int t = tid >> 6, dd = tid & 63;
  size_t n = (size_t)seq[t] * D_ + dd;
  g[tid] = (emb[n] + x1[n] + x2[n]) * (1.0f / 3.0f);
}

// ---------------------------------------------------------------- WMMA GEMM
// C[M,N] = act(A[M,K] @ W[K,N] + bias). K, N compile-time so the K/N tile
// loops fully unroll into straight-line v_wmma chains.
//
// Block = 8 waves. W is staged once per block into LDS as f16, transposed to
// [N][K] so a lane's B fragment (16 contiguous halves) is two ds_load_b128.
// Each wave owns one 16-row strip of C and all N/16 tiles; its A fragments
// (all K) are loaded with global_load_b128 and held in VGPRs, reused across
// the N tiles.
template<int K, int N, int RELU>
__global__ __launch_bounds__(256)
void gemm_wmma(const float* __restrict__ A, const float* __restrict__ W,
               const float* __restrict__ bias, float* __restrict__ C, int M) {
  constexpr int KSTEPS = K / 32;
  constexpr int NTILES = N / 16;
  __shared__ _Float16 Wlds[N * K];   // [n][k], f16

  // ---- stage + transpose + downconvert W (reads coalesced over n)
  for (int i = threadIdx.x; i < K * N; i += 256) {
    int k = i / N, n = i - k * N;
    Wlds[n * K + k] = (_Float16)W[i];
  }
  __syncthreads();

  int wave = threadIdx.x >> 5;
  int lane = threadIdx.x & 31;
  int hs   = lane >> 4;          // ISA K-split select (lanes 0-15 vs 16-31)
  int l16  = lane & 15;
  int strip = blockIdx.x * 8 + wave;
  if (strip * 16 >= M) return;   // no barriers after this point

  // ---- A fragments for the whole K range, kept in registers.
  // Lane's slice per 32-K step: floats [hs*16 .. hs*16+15].
  v16h afr[KSTEPS];
  const float4* ar =
      reinterpret_cast<const float4*>(A + (size_t)(strip * 16 + l16) * K) + hs * 4;
#pragma unroll
  for (int ks = 0; ks < KSTEPS; ++ks) {
    float4 a0 = ar[ks * 8 + 0];
    float4 a1 = ar[ks * 8 + 1];
    float4 a2 = ar[ks * 8 + 2];
    float4 a3 = ar[ks * 8 + 3];
    afr[ks] = (v16h){(_Float16)a0.x, (_Float16)a0.y, (_Float16)a0.z, (_Float16)a0.w,
                     (_Float16)a1.x, (_Float16)a1.y, (_Float16)a1.z, (_Float16)a1.w,
                     (_Float16)a2.x, (_Float16)a2.y, (_Float16)a2.z, (_Float16)a2.w,
                     (_Float16)a3.x, (_Float16)a3.y, (_Float16)a3.z, (_Float16)a3.w};
  }

  // ---- N tiles: B fragment = 16 contiguous halves from Wlds[col][k0+16*hs ..]
#pragma unroll
  for (int tn = 0; tn < NTILES; ++tn) {
    v8f acc = {};
    const _Float16* bp = Wlds + (size_t)(tn * 16 + l16) * K + hs * 16;
#pragma unroll
    for (int ks = 0; ks < KSTEPS; ++ks) {
      v16h bf = *reinterpret_cast<const v16h*>(bp + ks * 32);
      acc = wmma16x16x32(afr[ks], bf, acc);
    }
    float bv = bias[tn * 16 + l16];
#pragma unroll
    for (int r = 0; r < 8; ++r) {  // C/D: M = r + 8*hs, N = l16
      float o = acc[r] + bv;
      if (RELU) o = fmaxf(o, 0.0f);
      C[(size_t)(strip * 16 + r + 8 * hs) * N + tn * 16 + l16] = o;
    }
  }
}

// ---------------------------------------------------------------- blend + pos emb
__global__ void blend_pos(const int* __restrict__ seq, const int* __restrict__ lengths,
                          const float* __restrict__ sas, const float* __restrict__ glin,
                          const float* __restrict__ pos, float* __restrict__ x) {
  int tid = blockIdx.x * blockDim.x + threadIdx.x;
  if (tid >= T_ * D_) return;
  int t = tid >> 6, dd = tid & 63;
  int b = t / S_, s = t - b * S_;
  int len = lengths[b];
  float a = (len <= 10) ? 0.3f : ((len >= 50) ? 0.7f : 0.5f);
  size_t n = (size_t)seq[t] * D_ + dd;
  x[tid] = a * sas[n] + (1.0f - a) * glin[tid] + pos[s * D_ + dd];
}

// ---------------------------------------------------------------- attention
// dh = 32 == wave32: lane = feature dim. One wave per (b,h,q) row.
__global__ __launch_bounds__(32)
void attention_kernel(const float* __restrict__ q, const float* __restrict__ k,
                      const float* __restrict__ v, const int* __restrict__ seq,
                      float* __restrict__ out) {
  int idx = blockIdx.x;  // b*H*S + h*S + qi
  int qi = idx % S_;
  int h  = (idx / S_) % H_;
  int b  = idx / (S_ * H_);
  int lane = threadIdx.x;
  const float scale = 0.17677669529663687f;  // 1/sqrt(32)
  float qv = q[(size_t)(b * S_ + qi) * D_ + h * DH_ + lane];
  float sc[S_];
#pragma unroll
  for (int kk = 0; kk < S_; ++kk) {
    float kv = k[(size_t)(b * S_ + kk) * D_ + h * DH_ + lane];
    float dot = wave_sum32(qv * kv);
    bool valid = (kk <= qi) && (seq[b * S_ + kk] != 0);
    sc[kk] = valid ? dot * scale : -1.0e9f;
  }
  float mx = -1.0e30f;
#pragma unroll
  for (int kk = 0; kk < S_; ++kk) mx = fmaxf(mx, sc[kk]);
  float sum = 0.0f;
#pragma unroll
  for (int kk = 0; kk < S_; ++kk) { sc[kk] = __expf(sc[kk] - mx); sum += sc[kk]; }
  float inv = 1.0f / sum;
  float acc = 0.0f;
#pragma unroll
  for (int kk = 0; kk < S_; ++kk)
    acc += sc[kk] * v[(size_t)(b * S_ + kk) * D_ + h * DH_ + lane];
  out[(size_t)(b * S_ + qi) * D_ + h * DH_ + lane] = acc * inv;
}

// ---------------------------------------------------------------- residual + LayerNorm
// One wave per row (D=64 -> 2 elems/lane).
__global__ void add_layernorm(const float* __restrict__ x, const float* __restrict__ r,
                              const float* __restrict__ g, const float* __restrict__ bb,
                              float* __restrict__ outx, int rows) {
  int row = blockIdx.x * (blockDim.x >> 5) + (threadIdx.x >> 5);
  if (row >= rows) return;
  int lane = threadIdx.x & 31;
  const float* xr = x + (size_t)row * D_;
  const float* rr = r + (size_t)row * D_;
  float v0 = xr[lane] + rr[lane];
  float v1 = xr[lane + 32] + rr[lane + 32];
  float mu = wave_sum32(v0 + v1) * (1.0f / 64.0f);
  float d0 = v0 - mu, d1 = v1 - mu;
  float var = wave_sum32(d0 * d0 + d1 * d1) * (1.0f / 64.0f);
  float inv = rsqrtf(var + 1e-5f);
  float* orow = outx + (size_t)row * D_;
  orow[lane]      = d0 * inv * g[lane]      + bb[lane];
  orow[lane + 32] = d1 * inv * g[lane + 32] + bb[lane + 32];
}

// ---------------------------------------------------------------- final gather
__global__ void final_gather(const float* __restrict__ x, const int* __restrict__ lengths,
                             float* __restrict__ out) {
  int tid = blockIdx.x * blockDim.x + threadIdx.x;
  if (tid >= B_ * D_) return;
  int b = tid >> 6, dd = tid & 63;
  int t = b * S_ + lengths[b] - 1;
  out[tid] = x[(size_t)t * D_ + dd];
}

// ================================================================ host
struct BlkP {
  const float *wq, *bq, *wk, *bk, *wv, *bv, *wo, *bo;
  const float *ln1_g, *ln1_b, *w1, *b1, *w2, *b2, *ln2_g, *ln2_b;
};

extern "C" void kernel_launch(void* const* d_in, const int* in_sizes, int n_in,
                              void* d_out, int out_size, void* d_ws, size_t ws_size,
                              hipStream_t stream) {
  auto F = [&](int i) { return (const float*)d_in[i]; };
  auto I = [&](int i) { return (const int*)d_in[i]; };

  const int *seq, *lengths, *eidx;
  const float *ew, *sas, *gemb, *gw, *gb, *pos;
  BlkP blk[2];
  int E;
  long long N;

  if (in_sizes[0] == T_) {
    // Python dict insertion order: seq, lengths, edge_index, edge_weight,
    // sasrec_emb, gnn_emb, gnn_w, gnn_b, pos_emb, blocks[0..1] insertion order.
    seq = I(0); lengths = I(1); eidx = I(2); ew = F(3);
    sas = F(4); gemb = F(5); gw = F(6); gb = F(7); pos = F(8);
    for (int bx = 0; bx < 2; ++bx) {
      int o = 9 + bx * 16;
      blk[bx] = { F(o + 0), F(o + 1), F(o + 2), F(o + 3), F(o + 4), F(o + 5),
                  F(o + 6), F(o + 7), F(o + 8), F(o + 9), F(o + 10), F(o + 11),
                  F(o + 12), F(o + 13), F(o + 14), F(o + 15) };
    }
    E = in_sizes[3];
    N = (long long)in_sizes[4] / D_;
  } else {
    // jax pytree sorted-key order: edge_index, edge_weight, lengths,
    // blocks[0..1]{b1,b2,bk,bo,bq,bv,ln1_b,ln1_g,ln2_b,ln2_g,w1,w2,wk,wo,wq,wv},
    // gnn_b, gnn_emb, gnn_w, pos_emb, sasrec_emb, seq.
    eidx = I(0); ew = F(1); lengths = I(2);
    for (int bx = 0; bx < 2; ++bx) {
      int o = 3 + bx * 16;
      blk[bx].b1 = F(o + 0);  blk[bx].b2 = F(o + 1);  blk[bx].bk = F(o + 2);
      blk[bx].bo = F(o + 3);  blk[bx].bq = F(o + 4);  blk[bx].bv = F(o + 5);
      blk[bx].ln1_b = F(o + 6); blk[bx].ln1_g = F(o + 7);
      blk[bx].ln2_b = F(o + 8); blk[bx].ln2_g = F(o + 9);
      blk[bx].w1 = F(o + 10); blk[bx].w2 = F(o + 11); blk[bx].wk = F(o + 12);
      blk[bx].wo = F(o + 13); blk[bx].wq = F(o + 14); blk[bx].wv = F(o + 15);
    }
    gb = F(35); gemb = F(36); gw = F(37); pos = F(38); sas = F(39); seq = I(40);
    E = in_sizes[1];
    N = (long long)in_sizes[39] / D_;
  }

  // ---- workspace carve-up
  size_t N64 = (size_t)N * D_;
  float* p  = (float*)d_ws;
  float* x1 = p; p += N64;        // layer-1 segment sum (full graph)
  float* x2 = p; p += N64;        // layer-2 segment sum (full graph)
  float* gin = p; p += T_ * D_;   // gathered acc/3
  float* xb  = p; p += T_ * D_;   // transformer activations
  float* qb  = p; p += T_ * D_;
  float* kb  = p; p += T_ * D_;
  float* vb  = p; p += T_ * D_;
  float* ab  = p; p += T_ * D_;   // attention out
  float* tb  = p; p += T_ * D_;   // scratch (g_lin / proj out / ffn2 out)
  float* fb  = p; p += T_ * DFF_; // ffn hidden

  hipMemsetAsync(x1, 0, N64 * sizeof(float), stream);
  hipMemsetAsync(x2, 0, N64 * sizeof(float), stream);

  // ---- LightGCN: two edge-streaming scatter passes (atomic f32 adds, L2-resident)
  {
    long long total = (long long)E * 16;
    int blocks = (int)((total + 255) / 256);
    gnn_scatter<<<blocks, 256, 0, stream>>>(eidx, eidx + E, ew, gemb, x1, E);
    gnn_scatter<<<blocks, 256, 0, stream>>>(eidx, eidx + E, ew, x1, x2, E);
  }

  const int GB = (T_ / 16 + 7) / 8;  // GEMM blocks: 200 strips / 8 waves = 25

  // ---- GNN projection only at the 3200 gathered tokens, then blend + pos
  gather_acc<<<(T_ * D_ + 255) / 256, 256, 0, stream>>>(seq, gemb, x1, x2, gin);
  gemm_wmma<64, 64, 0><<<GB, 256, 0, stream>>>(gin, gw, gb, tb, T_);
  blend_pos<<<(T_ * D_ + 255) / 256, 256, 0, stream>>>(seq, lengths, sas, tb, pos, xb);

  // ---- transformer blocks
  for (int bx = 0; bx < 2; ++bx) {
    const BlkP& P = blk[bx];
    gemm_wmma<64, 64, 0><<<GB, 256, 0, stream>>>(xb, P.wq, P.bq, qb, T_);
    gemm_wmma<64, 64, 0><<<GB, 256, 0, stream>>>(xb, P.wk, P.bk, kb, T_);
    gemm_wmma<64, 64, 0><<<GB, 256, 0, stream>>>(xb, P.wv, P.bv, vb, T_);
    attention_kernel<<<B_ * H_ * S_, 32, 0, stream>>>(qb, kb, vb, seq, ab);
    gemm_wmma<64, 64, 0><<<GB, 256, 0, stream>>>(ab, P.wo, P.bo, tb, T_);
    add_layernorm<<<(T_ + 7) / 8, 256, 0, stream>>>(xb, tb, P.ln1_g, P.ln1_b, xb, T_);
    gemm_wmma<64, 256, 1><<<GB, 256, 0, stream>>>(xb, P.w1, P.b1, fb, T_);
    gemm_wmma<256, 64, 0><<<GB, 256, 0, stream>>>(fb, P.w2, P.b2, tb, T_);
    add_layernorm<<<(T_ + 7) / 8, 256, 0, stream>>>(xb, tb, P.ln2_g, P.ln2_b, xb, T_);
  }

  final_gather<<<(B_ * D_ + 63) / 64, 64, 0, stream>>>(xb, lengths, (float*)d_out);
}
